// OptimizedMultiHeadAttention_5119601017207
// MI455X (gfx1250) — compile-verified, matrix operand order checked by reading, not run on hardware
//
#include <hip/hip_runtime.h>

typedef __attribute__((ext_vector_type(16))) _Float16 v16h;
typedef __attribute__((ext_vector_type(8)))  _Float16 v8h;
typedef __attribute__((ext_vector_type(8)))  float    v8f;
typedef __attribute__((ext_vector_type(4)))  float    v4f;

#define BATCH 4
#define TSEQ  2048
#define CDIM  1024
#define NHEAD 16
#define DHEAD 64
#define F3    3072
#define MROWS (BATCH * TSEQ)                            // 8192
#define QKV_SZ ((size_t)BATCH * NHEAD * TSEQ * DHEAD)   // 8388608 elems per tensor

__device__ __forceinline__ v8f wmma16(v16h a, v16h b, v8f c) {
    return __builtin_amdgcn_wmma_f32_16x16x32_f16(
        /*neg_a=*/false, a, /*neg_b=*/false, b,
        /*c_mod=*/(short)0, c, /*reuse_a=*/false, /*reuse_b=*/false);
}

// A/B fragment layout (16-bit, 16x32): lane L holds row (L&15);
// halves 0..7  = K[kO .. kO+7],  halves 8..15 = K[kO+16 .. kO+23], kO = (L>>4)*8.
// Caller passes p = rowBase + kO.
__device__ __forceinline__ v16h frag_from_f16(const _Float16* __restrict__ p) {
    v8h lo = *reinterpret_cast<const v8h*>(p);
    v8h hi = *reinterpret_cast<const v8h*>(p + 16);
    v16h r;
#pragma unroll
    for (int i = 0; i < 8; ++i) { r[i] = lo[i]; r[8 + i] = hi[i]; }
    return r;
}

// ---------------------------------------------------------------------------
// Kernel 0: one-shot fp32 -> f16 conversion (bandwidth bound).
// Removes all per-tile v_cvt work from the GEMM hot loops.
// ---------------------------------------------------------------------------
__global__ __launch_bounds__(256) void cvt_f16_kernel(
    const float* __restrict__ src, _Float16* __restrict__ dst, int n8) {
    int i = blockIdx.x * blockDim.x + threadIdx.x;
    if (i >= n8) return;
    const v4f* s = reinterpret_cast<const v4f*>(src) + 2 * (size_t)i;
    v4f f0 = s[0], f1 = s[1];
    v8h h;
#pragma unroll
    for (int j = 0; j < 4; ++j) {
        h[j]     = (_Float16)f0[j];
        h[4 + j] = (_Float16)f1[j];
    }
    reinterpret_cast<v8h*>(dst)[i] = h;
}

// ---------------------------------------------------------------------------
// Kernel 1: QKV projection. qkv[m, n] = sum_k x[m,k] * w_attn[n,k] (f16 in)
// Block: 256 thr = 8 waves, tile 128(M) x 128(N); wave = 32(M) x 64(N).
// Writes q,k as f16 [B,H,T,D]; v transposed as f16 [B,H,D,T].
// ---------------------------------------------------------------------------
__global__ __launch_bounds__(256) void qkv_gemm_kernel(
    const _Float16* __restrict__ x, const _Float16* __restrict__ w,
    _Float16* __restrict__ qkv) {
    const int lane = threadIdx.x & 31;
    const int wave = threadIdx.x >> 5;
    const int wm = wave & 3, wn = wave >> 2;          // 4 (M) x 2 (N) waves
    const int Mb = blockIdx.y * 128 + wm * 32;
    const int Nb = blockIdx.x * 128 + wn * 64;
    const int r16 = lane & 15;
    const int g8  = (lane >> 4) * 8;

    v8f acc[2][4] = {};
    const _Float16* xr0 = x + (size_t)(Mb + r16) * CDIM + g8;
    const _Float16* xr1 = x + (size_t)(Mb + 16 + r16) * CDIM + g8;
    const _Float16* wr[4];
#pragma unroll
    for (int ni = 0; ni < 4; ++ni)
        wr[ni] = w + (size_t)(Nb + ni * 16 + r16) * CDIM + g8;

    for (int kb = 0; kb < CDIM; kb += 32) {
        // global_prefetch_b8 one K-iteration ahead (streaming rows)
        if (kb + 32 < CDIM) {
            __builtin_prefetch(xr0 + kb + 32, 0, 1);
            __builtin_prefetch(xr1 + kb + 32, 0, 1);
            __builtin_prefetch(wr[0] + kb + 32, 0, 1);
            __builtin_prefetch(wr[2] + kb + 32, 0, 1);
        }
        v16h a0 = frag_from_f16(xr0 + kb);
        v16h a1 = frag_from_f16(xr1 + kb);
#pragma unroll
        for (int ni = 0; ni < 4; ++ni) {
            v16h bf = frag_from_f16(wr[ni] + kb);
            acc[0][ni] = wmma16(a0, bf, acc[0][ni]);
            acc[1][ni] = wmma16(a1, bf, acc[1][ni]);
        }
    }
    // epilogue: D layout -> lane holds col (lane&15), rows (lane>>4)*8 + e
#pragma unroll
    for (int mi = 0; mi < 2; ++mi) {
#pragma unroll
        for (int ni = 0; ni < 4; ++ni) {
#pragma unroll
            for (int e = 0; e < 8; ++e) {
                int m = Mb + mi * 16 + g8 + e;
                int n = Nb + ni * 16 + r16;
                int which = n >> 10;          // 0=q 1=k 2=v
                int c = n & 1023;
                int h = c >> 6, d = c & 63;
                int b_ = m >> 11, t = m & 2047;
                size_t off;
                if (which == 2)   // v: [b,h,d,t]
                    off = 2 * QKV_SZ + (((size_t)(b_ * NHEAD + h) * DHEAD + d) * TSEQ + t);
                else              // q,k: [b,h,t,d]
                    off = (size_t)which * QKV_SZ + (((size_t)(b_ * NHEAD + h) * TSEQ + t) * DHEAD + d);
                qkv[off] = (_Float16)acc[mi][ni][e];
            }
        }
    }
}

// ---------------------------------------------------------------------------
// Kernel 2: causal flash attention. One wave owns 16 query rows; block of
// 8 waves covers 128 rows of one (b,h). Key tiles of 64.
// ---------------------------------------------------------------------------
__global__ __launch_bounds__(256) void attn_kernel(
    const _Float16* __restrict__ q, const _Float16* __restrict__ k,
    const _Float16* __restrict__ v, _Float16* __restrict__ yout) {
    __shared__ _Float16 pstage[8][16 * 72];   // per-wave 16x64 P tile, stride 72 (16B aligned)
    const int lane = threadIdx.x & 31;
    const int wave = threadIdx.x >> 5;
    const int bh   = blockIdx.y;              // b*16 + h
    const int qrow0 = blockIdx.x * 128 + wave * 16;
    const int r16 = lane & 15;
    const int g8  = (lane >> 4) * 8;
    const float scale = 0.125f;               // 1/sqrt(64)

    const _Float16* qp = q + (size_t)bh * TSEQ * DHEAD;
    const _Float16* kp = k + (size_t)bh * TSEQ * DHEAD;
    const _Float16* vp = v + (size_t)bh * DHEAD * TSEQ;

    // Q fragments: 16 rows x D=64 -> two K=32 frags
    v16h qf0 = frag_from_f16(qp + (size_t)(qrow0 + r16) * DHEAD + g8);
    v16h qf1 = frag_from_f16(qp + (size_t)(qrow0 + r16) * DHEAD + 32 + g8);

    float m_run[8], l_run[8];
#pragma unroll
    for (int e = 0; e < 8; ++e) { m_run[e] = -1e30f; l_run[e] = 0.0f; }
    v8f o[4] = {};

    _Float16* pl = &pstage[wave][0];
    const int ktiles = (qrow0 + 16 + 63) >> 6;   // tiles covering keys [0, qrow0+15]

    for (int kt = 0; kt < ktiles; ++kt) {
        const int kbase = kt * 64;
        // ---- S = Q * K^T (16 x 64), f32 accum ----
        v8f s[4] = {};
#pragma unroll
        for (int nj = 0; nj < 4; ++nj) {
            const _Float16* kr = kp + (size_t)(kbase + nj * 16 + r16) * DHEAD + g8;
            v16h b0 = frag_from_f16(kr);
            v16h b1 = frag_from_f16(kr + 32);
            s[nj] = wmma16(qf0, b0, s[nj]);
            s[nj] = wmma16(qf1, b1, s[nj]);
        }
        // ---- scale + causal mask ----
        const int rbase = qrow0 + g8;
#pragma unroll
        for (int nj = 0; nj < 4; ++nj) {
            const int col = kbase + nj * 16 + r16;
#pragma unroll
            for (int e = 0; e < 8; ++e) {
                float val = s[nj][e] * scale;
                if (col > rbase + e) val = -1e30f;
                s[nj][e] = val;
            }
        }
        // ---- online softmax (row = rbase+e lives in 16 lanes of one half) ----
#pragma unroll
        for (int e = 0; e < 8; ++e) {
            float mx = fmaxf(fmaxf(s[0][e], s[1][e]), fmaxf(s[2][e], s[3][e]));
#pragma unroll
            for (int off = 1; off < 16; off <<= 1)
                mx = fmaxf(mx, __shfl_xor(mx, off, 32));
            float mnew = fmaxf(m_run[e], mx);
            float alpha = __expf(m_run[e] - mnew);
            m_run[e] = mnew;
            l_run[e] *= alpha;
#pragma unroll
            for (int dj = 0; dj < 4; ++dj) o[dj][e] *= alpha;
            float se = 0.0f;
#pragma unroll
            for (int nj = 0; nj < 4; ++nj) {
                float pij = __expf(s[nj][e] - mnew);
                s[nj][e] = pij;
                se += pij;
            }
#pragma unroll
            for (int off = 1; off < 16; off <<= 1)
                se += __shfl_xor(se, off, 32);
            l_run[e] += se;
        }
        // ---- stage P (16x64) to LDS to re-layout into an A operand ----
#pragma unroll
        for (int nj = 0; nj < 4; ++nj)
#pragma unroll
            for (int e = 0; e < 8; ++e)
                pl[(g8 + e) * 72 + nj * 16 + r16] = (_Float16)s[nj][e];
        asm volatile("s_wait_dscnt 0" ::: "memory");
        v16h pa0 = frag_from_f16(pl + r16 * 72 + g8);        // keys 0..31
        v16h pa1 = frag_from_f16(pl + r16 * 72 + 32 + g8);   // keys 32..63
        // ---- O += P * V  (V stored [d][t] -> B operand, contiguous keys) ----
#pragma unroll
        for (int dj = 0; dj < 4; ++dj) {
            const _Float16* vr = vp + (size_t)(dj * 16 + r16) * TSEQ + kbase + g8;
            v16h vb0 = frag_from_f16(vr);
            v16h vb1 = frag_from_f16(vr + 32);
            o[dj] = wmma16(pa0, vb0, o[dj]);
            o[dj] = wmma16(pa1, vb1, o[dj]);
        }
    }
    // ---- normalize + store y as f16 [B,T,C] with c = h*64+d ----
    const int b_ = bh >> 4, h = bh & 15;
#pragma unroll
    for (int dj = 0; dj < 4; ++dj) {
#pragma unroll
        for (int e = 0; e < 8; ++e) {
            int t = qrow0 + g8 + e;
            int d = dj * 16 + r16;
            float val = o[dj][e] / l_run[e];
            yout[((size_t)(b_ * TSEQ + t)) * CDIM + h * DHEAD + d] = (_Float16)val;
        }
    }
}

// ---------------------------------------------------------------------------
// Kernel 3: output projection. out[m,n] = sum_k y[m,k]*w_proj[n,k] + b[n]
// ---------------------------------------------------------------------------
__global__ __launch_bounds__(256) void proj_gemm_kernel(
    const _Float16* __restrict__ y, const _Float16* __restrict__ w,
    const float* __restrict__ bias, float* __restrict__ out) {
    const int lane = threadIdx.x & 31;
    const int wave = threadIdx.x >> 5;
    const int wm = wave & 3, wn = wave >> 2;
    const int Mb = blockIdx.y * 128 + wm * 32;
    const int Nb = blockIdx.x * 128 + wn * 64;
    const int r16 = lane & 15;
    const int g8  = (lane >> 4) * 8;

    v8f acc[2][4] = {};
    const _Float16* yr0 = y + (size_t)(Mb + r16) * CDIM + g8;
    const _Float16* yr1 = y + (size_t)(Mb + 16 + r16) * CDIM + g8;
    const _Float16* wr[4];
#pragma unroll
    for (int ni = 0; ni < 4; ++ni)
        wr[ni] = w + (size_t)(Nb + ni * 16 + r16) * CDIM + g8;

    for (int kb = 0; kb < CDIM; kb += 32) {
        if (kb + 32 < CDIM) {
            __builtin_prefetch(yr0 + kb + 32, 0, 1);
            __builtin_prefetch(wr[0] + kb + 32, 0, 1);
        }
        v16h a0 = frag_from_f16(yr0 + kb);
        v16h a1 = frag_from_f16(yr1 + kb);
#pragma unroll
        for (int ni = 0; ni < 4; ++ni) {
            v16h bf = frag_from_f16(wr[ni] + kb);
            acc[0][ni] = wmma16(a0, bf, acc[0][ni]);
            acc[1][ni] = wmma16(a1, bf, acc[1][ni]);
        }
    }
#pragma unroll
    for (int mi = 0; mi < 2; ++mi) {
#pragma unroll
        for (int ni = 0; ni < 4; ++ni) {
#pragma unroll
            for (int e = 0; e < 8; ++e) {
                int m = Mb + mi * 16 + g8 + e;
                int n = Nb + ni * 16 + r16;
                out[(size_t)m * CDIM + n] = acc[mi][ni][e] + bias[n];
            }
        }
    }
}

extern "C" void kernel_launch(void* const* d_in, const int* in_sizes, int n_in,
                              void* d_out, int out_size, void* d_ws, size_t ws_size,
                              hipStream_t stream) {
    const float* x      = (const float*)d_in[0];
    const float* w_attn = (const float*)d_in[1];
    const float* w_proj = (const float*)d_in[2];
    const float* b_proj = (const float*)d_in[3];
    float* out = (float*)d_out;

    const size_t X_SZ  = (size_t)MROWS * CDIM;   // 8388608
    const size_t WA_SZ = (size_t)F3 * CDIM;      // 3145728
    const size_t WP_SZ = (size_t)CDIM * CDIM;    // 1048576

    _Float16* xh  = (_Float16*)d_ws;
    _Float16* wah = xh + X_SZ;
    _Float16* wph = wah + WA_SZ;
    _Float16* qkv = wph + WP_SZ;                 // 3 * QKV_SZ halves (48 MB)
    _Float16* yh  = qkv + 3 * QKV_SZ;            // X_SZ halves (16 MB)

    // fp32 -> f16 pre-pass (removes all cvt work from GEMM hot loops)
    cvt_f16_kernel<<<dim3((int)(X_SZ / 8 + 255) / 256), 256, 0, stream>>>(x, xh, (int)(X_SZ / 8));
    cvt_f16_kernel<<<dim3((int)(WA_SZ / 8 + 255) / 256), 256, 0, stream>>>(w_attn, wah, (int)(WA_SZ / 8));
    cvt_f16_kernel<<<dim3((int)(WP_SZ / 8 + 255) / 256), 256, 0, stream>>>(w_proj, wph, (int)(WP_SZ / 8));

    qkv_gemm_kernel<<<dim3(F3 / 128, MROWS / 128), 256, 0, stream>>>(xh, wah, qkv);
    attn_kernel<<<dim3(TSEQ / 128, BATCH * NHEAD), 256, 0, stream>>>(
        qkv, qkv + QKV_SZ, qkv + 2 * QKV_SZ, yh);
    proj_gemm_kernel<<<dim3(CDIM / 128, MROWS / 128), 256, 0, stream>>>(
        yh, wph, b_proj, out);
}
